// ThinkingRefiner_50525995270579
// MI455X (gfx1250) — matrix operand order repaired and checked by reading, not compiled, not run on hardware
//
#include <hip/hip_runtime.h>
#include <hip/hip_bf16.h>
#include <math.h>

typedef __attribute__((ext_vector_type(16))) _Float16 v16h;
typedef __attribute__((ext_vector_type(8)))  float    v8f;
typedef __attribute__((ext_vector_type(4)))  _Float16 v4h;
typedef __attribute__((ext_vector_type(4)))  float    v4f;

#define HDIM 1024
#define MT   64            // rows per workgroup
#define XS   1028          // f32 row stride in floats (4112B: 16B aligned, dwords%64==4)
#define NKB  32            // K blocks of 32 (1024/32)
#define AROW 80            // bytes per row inside one K-block (64B data + 16B pad)
#define ABLK 5152          // bytes per K-block (64 rows * 80B + 32B pad)

union Frag {
    v16h  v;
    uint4 q[2];
};
union Pack8 {
    uint4     q;
    _Float16  h[8];
};

__device__ __forceinline__ float waveReduceSum(float v) {
#pragma unroll
    for (int m = 16; m >= 1; m >>= 1) v += __shfl_xor(v, m, 32);
    return v;
}

__device__ __forceinline__ float gelu_exact(float v) {
    return 0.5f * v * (1.f + erff(v * 0.70710678118654752f));
}

// ---------------------------------------------------------------------------
// Kernel 1: convert W (f32, [O=1024][H=1024]) into f16 WMMA-B fragments.
// Fragment f = nTile*NKB + kb covers B[k][n] = W[n*H + k] for
//   n in [nTile*16, nTile*16+16), k in [kb*32, kb*32+32).
// Per-lane layout (wave32, B 32x16 f16): lane L -> n = nTile*16 + (L&15),
//   k0 = kb*32 + (L>>4)*16, 16 consecutive K values. Stored chunked so GEMM
//   loads are two fully coalesced global_load_b128 per fragment.
// ---------------------------------------------------------------------------
__global__ __launch_bounds__(256) void wpack_kernel(const float* __restrict__ W,
                                                    _Float16* __restrict__ wq) {
    const int t    = blockIdx.x * 256 + threadIdx.x;   // 65536 threads total
    const int f    = t >> 5;
    const int lane = t & 31;
    const int nTile = f >> 5;          // 0..63
    const int kb    = f & 31;          // 0..31
    const int n  = nTile * 16 + (lane & 15);
    const int k0 = kb * 32 + (lane >> 4) * 16;

    const float4* s4 = (const float4*)(W + (size_t)n * HDIM + k0);
    float4 f0 = s4[0], f1 = s4[1], f2 = s4[2], f3 = s4[3];

    Pack8 c0, c1;
    c0.h[0] = (_Float16)f0.x; c0.h[1] = (_Float16)f0.y;
    c0.h[2] = (_Float16)f0.z; c0.h[3] = (_Float16)f0.w;
    c0.h[4] = (_Float16)f1.x; c0.h[5] = (_Float16)f1.y;
    c0.h[6] = (_Float16)f1.z; c0.h[7] = (_Float16)f1.w;
    c1.h[0] = (_Float16)f2.x; c1.h[1] = (_Float16)f2.y;
    c1.h[2] = (_Float16)f2.z; c1.h[3] = (_Float16)f2.w;
    c1.h[4] = (_Float16)f3.x; c1.h[5] = (_Float16)f3.y;
    c1.h[6] = (_Float16)f3.z; c1.h[7] = (_Float16)f3.w;

    uint4* dst = (uint4*)(wq + (size_t)f * 512);
    dst[lane]      = c0.q;
    dst[32 + lane] = c1.q;
}

// ---------------------------------------------------------------------------
// Kernel 2: fused LayerNorm -> GEMM(WMMA f16) -> bias+GELU -> residual ->
//           RMSNorm -> masked select. One WG per 64 rows, 512 threads.
// ---------------------------------------------------------------------------
__global__ __launch_bounds__(512) void fused_kernel(
    const float* __restrict__ x, const int* __restrict__ imask,
    const float* __restrict__ ln_w, const float* __restrict__ ln_b,
    const _Float16* __restrict__ wq, const float* __restrict__ bias,
    const float* __restrict__ alphap, const float* __restrict__ rms_w,
    float* __restrict__ out) {

    // Union region: phase 1/2 use K-block-major f16 h (<=161KB),
    // phase 3 uses 64 x XS f32 (263168B, dominates).
    __shared__ __align__(16) unsigned char lds_raw[MT * XS * 4];
    __shared__ float rs_arr[MT];
    __shared__ int   msk_arr[MT];

    float* xf = (float*)lds_raw;    // 64 x XS floats (phase 3)

    const int tid   = threadIdx.x;
    const int wv    = tid >> 5;     // 0..15
    const int lane  = tid & 31;
    const int halfL = lane >> 4;
    const int ml    = lane & 15;
    const size_t rowBase = (size_t)blockIdx.x * MT;

    // ---- Phase 1: LayerNorm 4 rows per wave, h -> LDS (K-block-major f16) --
#pragma unroll
    for (int rr = 0; rr < 4; ++rr) {
        const int r = wv * 4 + rr;
        const float4* xrow = (const float4*)(x + (rowBase + r) * HDIM);
        float4 xv[8];
        float s1 = 0.f, s2 = 0.f;
#pragma unroll
        for (int k = 0; k < 8; ++k) {
            xv[k] = xrow[lane + k * 32];
            s1 += xv[k].x + xv[k].y + xv[k].z + xv[k].w;
            s2 += xv[k].x * xv[k].x + xv[k].y * xv[k].y +
                  xv[k].z * xv[k].z + xv[k].w * xv[k].w;
        }
        s1 = waveReduceSum(s1);
        s2 = waveReduceSum(s2);
        const float mu  = s1 * (1.f / HDIM);
        const float var = s2 * (1.f / HDIM) - mu * mu;
        const float rst = rsqrtf(var + 1e-5f);
#pragma unroll
        for (int k = 0; k < 8; ++k) {
            const int c4  = lane + k * 32;
            const int col = c4 * 4;              // global K index of 4 halves
            float4 w4 = ((const float4*)ln_w)[c4];
            float4 b4 = ((const float4*)ln_b)[c4];
            v4h hv;
            hv[0] = (_Float16)((xv[k].x - mu) * rst * w4.x + b4.x);
            hv[1] = (_Float16)((xv[k].y - mu) * rst * w4.y + b4.y);
            hv[2] = (_Float16)((xv[k].z - mu) * rst * w4.z + b4.z);
            hv[3] = (_Float16)((xv[k].w - mu) * rst * w4.w + b4.w);
            // block (col>>5), row r, intra-block halves (col&31)
            *(v4h*)(lds_raw + (col >> 5) * ABLK + r * AROW + (col & 31) * 2) = hv;
        }
    }
    __syncthreads();

    // ---- Phase 2: GEMM. Wave wv owns cols [wv*64, wv*64+64). ----
    v8f acc[4][4];
    const v8f vzero = {0.f, 0.f, 0.f, 0.f, 0.f, 0.f, 0.f, 0.f};
#pragma unroll
    for (int mt = 0; mt < 4; ++mt)
#pragma unroll
        for (int nt = 0; nt < 4; ++nt) acc[mt][nt] = vzero;

    // Single per-lane LDS base (integer offset keeps addrspace(3) => ds ops);
    // all 8 A loads per K-step are ds_load_b128 with immediate offsets.
    unsigned abase = (unsigned)(ml * AROW + halfL * 32);

    for (int kb = 0; kb < NKB; ++kb) {
        Frag a[4];
#pragma unroll
        for (int mt = 0; mt < 4; ++mt) {
            // halves[0..15] = K kb*32+half*16.. (16 consecutive)
            a[mt].q[0] = *(const uint4*)(lds_raw + abase + mt * 16 * AROW);
            a[mt].q[1] = *(const uint4*)(lds_raw + abase + mt * 16 * AROW + 16);
        }
        abase += ABLK;
        Frag bf[4];
#pragma unroll
        for (int nt = 0; nt < 4; ++nt) {
            const uint4* bp =
                (const uint4*)(wq + ((size_t)(wv * 4 + nt) * NKB + kb) * 512);
            bf[nt].q[0] = bp[lane];
            bf[nt].q[1] = bp[32 + lane];
        }
#pragma unroll
        for (int mt = 0; mt < 4; ++mt)
#pragma unroll
            for (int nt = 0; nt < 4; ++nt)
                acc[mt][nt] = __builtin_amdgcn_wmma_f32_16x16x32_f16(
                    false, a[mt].v, false, bf[nt].v, (short)0, acc[mt][nt],
                    false, false);
    }
    __syncthreads();  // all reads of h done; LDS reused as xf

    // ---- Phase 3a: dump acc + bias into LDS (conflict-free scattered stores)
    // Element (m,n) owned by exactly one (wave,lane,mt,nt,j): no races.
#pragma unroll
    for (int nt = 0; nt < 4; ++nt) {
        const int n  = wv * 64 + nt * 16 + ml;
        const float bn = bias[n];
#pragma unroll
        for (int mt = 0; mt < 4; ++mt) {
#pragma unroll
            for (int j = 0; j < 8; ++j) {
                const int m = mt * 16 + halfL * 8 + j;  // C/D layout: vgpr j
                xf[m * XS + n] = acc[mt][nt][j] + bn;
            }
        }
    }
    __syncthreads();

    // ---- Phase 3b: rolled GELU + alpha-residual pass.
    // LDS float4 read (pre-activation) + coalesced global float4 read of x.
    const float alpha = *alphap;
    for (int idx = tid; idx < MT * (HDIM / 4); idx += 512) {
        const int r = idx >> 8, c4 = idx & 255;
        float* fp = xf + r * XS + c4 * 4;
        float4 vv = *(float4*)fp;
        float4 xx = ((const float4*)(x + (rowBase + r) * HDIM))[c4];
        float4 f;
        f.x = xx.x + alpha * gelu_exact(vv.x);
        f.y = xx.y + alpha * gelu_exact(vv.y);
        f.z = xx.z + alpha * gelu_exact(vv.z);
        f.w = xx.w + alpha * gelu_exact(vv.w);
        *(float4*)fp = f;
    }
    __syncthreads();

    // ---- Phase 3c: per-row sum of squares -> rsqrt, plus mask bit ----
#pragma unroll
    for (int rr = 0; rr < 4; ++rr) {
        const int r = wv * 4 + rr;
        float s = 0.f;
#pragma unroll
        for (int k = 0; k < 32; ++k) {
            float f = xf[r * XS + lane + k * 32];
            s += f * f;
        }
        s = waveReduceSum(s);
        if (lane == 0) {
            rs_arr[r]  = rsqrtf(s * (1.f / HDIM) + 1e-6f);
            msk_arr[r] = (imask[rowBase + r] == 2) ? 1 : 0;
        }
    }
    __syncthreads();

    // ---- Phase 3d: RMS scale + masked select, nontemporal coalesced store --
    for (int idx = tid; idx < MT * (HDIM / 4); idx += 512) {
        const int r = idx >> 8, c4 = idx & 255;
        const size_t rg = rowBase + r;
        v4f o;
        if (msk_arr[r]) {
            const float rsv = rs_arr[r];
            float4 w4 = ((const float4*)rms_w)[c4];
            const float* fp = xf + r * XS + c4 * 4;
            o.x = fp[0] * rsv * w4.x;
            o.y = fp[1] * rsv * w4.y;
            o.z = fp[2] * rsv * w4.z;
            o.w = fp[3] * rsv * w4.w;
        } else {
            float4 xx = ((const float4*)(x + rg * HDIM))[c4];
            o.x = xx.x; o.y = xx.y; o.z = xx.z; o.w = xx.w;
        }
        __builtin_nontemporal_store(o, (v4f*)(out + rg * HDIM) + c4);
    }
}

extern "C" void kernel_launch(void* const* d_in, const int* in_sizes, int n_in,
                              void* d_out, int out_size, void* d_ws, size_t ws_size,
                              hipStream_t stream) {
    (void)n_in; (void)out_size; (void)ws_size;
    const float* x      = (const float*)d_in[0];
    const int*   imask  = (const int*)d_in[1];
    const float* ln_w   = (const float*)d_in[2];
    const float* ln_b   = (const float*)d_in[3];
    const float* W      = (const float*)d_in[4];
    const float* bias   = (const float*)d_in[5];
    const float* alphap = (const float*)d_in[6];
    const float* rms_w  = (const float*)d_in[7];
    float*       out    = (float*)d_out;
    _Float16*    wq     = (_Float16*)d_ws;   // 1024*1024 f16 = 2 MB

    const int rows = in_sizes[0] / HDIM;     // 65536

    // 2048 fragments * 32 lanes = 65536 threads
    wpack_kernel<<<256, 256, 0, stream>>>(W, wq);
    fused_kernel<<<rows / MT, 512, 0, stream>>>(x, imask, ln_w, ln_b, wq, bias,
                                                alphap, rms_w, out);
}
